// DeepSeekV2MoeLayer_16690242913252
// MI455X (gfx1250) — matrix operand order repaired and checked by reading, not compiled, not run on hardware
//
#include <hip/hip_runtime.h>
#include <hip/hip_bf16.h>

#define HIDDEN 2048
#define INTER  1408
#define NEXP   8
#define TOKENS 2048
#define KP32   20   // padded LDS row pitch in u32 (40 bf16 = 80B: 16B-aligned, conflict-free)

typedef __bf16 bf16;
typedef __bf16 v16bf __attribute__((ext_vector_type(16)));
typedef __bf16 v8bf  __attribute__((ext_vector_type(8)));
typedef float  v8f   __attribute__((ext_vector_type(8)));

// Concatenate two contiguous 16B LDS chunks into one 16-element bf16 fragment.
#define FRAGP(ptr, o0, o1)                                                     \
  __builtin_shufflevector(*reinterpret_cast<const v8bf*>((ptr) + (o0)),        \
                          *reinterpret_cast<const v8bf*>((ptr) + (o1)),        \
                          0, 1, 2, 3, 4, 5, 6, 7, 8, 9, 10, 11, 12, 13, 14, 15)

__device__ __forceinline__ unsigned pack_bf16(float a, float b) {
  union { bf16 h[2]; unsigned u; } p;
  p.h[0] = (bf16)a;  // K even -> low half
  p.h[1] = (bf16)b;  // K odd  -> high half
  return p.u;
}

// ---------------------------------------------------------------------------
// Kernel 0: zero the output (harness poisons d_out with 0xAA)
// ---------------------------------------------------------------------------
__global__ void zero_out_kernel(float* __restrict__ out, int n) {
  int i = blockIdx.x * blockDim.x + threadIdx.x;
  int stride = gridDim.x * blockDim.x;
  for (; i < n; i += stride) out[i] = 0.0f;
}

// ---------------------------------------------------------------------------
// Kernel 1: router — one wave(32) per token. Lane-parallel dot products over
// HIDDEN, shuffle reduction, softmax(8), top-2, renormalize.
// ---------------------------------------------------------------------------
__global__ void __launch_bounds__(128) router_kernel(
    const float* __restrict__ x, const float* __restrict__ wr,
    int* __restrict__ topk_idx, float* __restrict__ topk_w) {
  const int wave = threadIdx.x >> 5;
  const int lane = threadIdx.x & 31;
  const int t = blockIdx.x * 4 + wave;
  if (t >= TOKENS) return;

  float acc[NEXP];
#pragma unroll
  for (int e = 0; e < NEXP; ++e) acc[e] = 0.0f;

  const float* xr = x + (size_t)t * HIDDEN;
  for (int h = lane; h < HIDDEN; h += 32) {
    float xv = xr[h];
#pragma unroll
    for (int e = 0; e < NEXP; ++e) acc[e] += xv * wr[h * NEXP + e];
  }
#pragma unroll
  for (int off = 16; off > 0; off >>= 1) {
#pragma unroll
    for (int e = 0; e < NEXP; ++e) acc[e] += __shfl_xor(acc[e], off, 32);
  }

  if (lane == 0) {
    float mx = acc[0];
#pragma unroll
    for (int e = 1; e < NEXP; ++e) mx = fmaxf(mx, acc[e]);
    float p[NEXP];
    float s = 0.0f;
#pragma unroll
    for (int e = 0; e < NEXP; ++e) { p[e] = __expf(acc[e] - mx); s += p[e]; }
    float inv = 1.0f / s;
#pragma unroll
    for (int e = 0; e < NEXP; ++e) p[e] *= inv;
    int i0 = 0;
#pragma unroll
    for (int e = 1; e < NEXP; ++e) if (p[e] > p[i0]) i0 = e;
    int i1 = -1;
#pragma unroll
    for (int e = 0; e < NEXP; ++e)
      if (e != i0 && (i1 < 0 || p[e] > p[i1])) i1 = e;
    float w0 = p[i0], w1 = p[i1];
    float isw = 1.0f / (w0 + w1);
    topk_idx[t * 2 + 0] = i0;
    topk_idx[t * 2 + 1] = i1;
    topk_w[t * 2 + 0] = w0 * isw;
    topk_w[t * 2 + 1] = w1 * isw;
  }
}

// ---------------------------------------------------------------------------
// Kernel 2: deterministic per-expert token compaction (ballot prefix).
// Lists padded to a multiple of 32 rows with (token 0, weight 0) dummies.
// ---------------------------------------------------------------------------
__global__ void build_lists_kernel(
    const int* __restrict__ topk_idx, const float* __restrict__ topk_w,
    int* __restrict__ token_list, float* __restrict__ weight_list,
    int* __restrict__ padded_counts) {
  const int e = blockIdx.x;
  const int lane = threadIdx.x;  // blockDim.x == 32
  int cnt = 0;
  for (int base = 0; base < TOKENS; base += 32) {
    int t = base + lane;
    int hit = 0;
    float w = 0.0f;
    int i0 = topk_idx[t * 2 + 0];
    int i1 = topk_idx[t * 2 + 1];
    if (i0 == e)      { hit = 1; w = topk_w[t * 2 + 0]; }
    else if (i1 == e) { hit = 1; w = topk_w[t * 2 + 1]; }
    unsigned mask = (unsigned)__ballot(hit);
    if (hit) {
      int pos = cnt + __popc(mask & ((1u << lane) - 1u));
      token_list[e * TOKENS + pos] = t;
      weight_list[e * TOKENS + pos] = w;
    }
    cnt += __popc(mask);
  }
  int padded = (cnt + 31) & ~31;
  for (int i = cnt + lane; i < padded; i += 32) {
    token_list[e * TOKENS + i] = 0;
    weight_list[e * TOKENS + i] = 0.0f;
  }
  if (lane == 0) padded_counts[e] = padded;
}

// ---------------------------------------------------------------------------
// WMMA fragment layouts (CDNA5, wave32; A/B use this code's contiguous k order):
//   A 16x32 bf16: lane L: M=L&15; elems [0..15] = K kb..kb+15, kb = 16*(L>>4)
//                 -> 2 contiguous 16B chunks
//   B 32x16 bf16: lane L: N=L&15; elems [0..15] = K kb..kb+15, kb = 16*(L>>4)
//                 -> 2 contiguous 16B chunks when stored [n][k]
//   C/D 16x16 f32: elem j: M = j + 8*(lane>>4), N = lane&15
// ---------------------------------------------------------------------------

// Kernel 3: grouped gate/up GEMM + SiLU epilogue -> compact bf16 hbuf.
// Block = 128 threads (4 waves). Tile: 32 tokens x 128 INTER cols.
// All staging pointers hoisted to registers; B pointers walk K continuously.
__global__ void __launch_bounds__(128) gateup_kernel(
    const float* __restrict__ x, const float* __restrict__ wg,
    const float* __restrict__ wu, const int* __restrict__ token_list,
    const float* __restrict__ weight_list,
    const int* __restrict__ padded_counts, bf16* __restrict__ hbuf) {
  const int e = blockIdx.z;
  const int rowTile = blockIdx.y;
  const int nBase = blockIdx.x * 128;
  const int pcnt = padded_counts[e];
  if (rowTile * 32 >= pcnt) return;
  int slotBase = 0;
  for (int i = 0; i < e; ++i) slotBase += padded_counts[i];
  slotBase += rowTile * 32;

  __shared__ unsigned As32[32 * KP32];    // A  [m][k]  bf16-packed
  __shared__ unsigned Bg32[128 * KP32];   // Bg [n][k]  bf16-packed (transposed)
  __shared__ unsigned Bu32[128 * KP32];   // Bu [n][k]
  __shared__ float wts[32];

  const int tid  = threadIdx.x;
  const int lane = tid & 31;
  const int wave = tid >> 5;

  // ---- staging roles (fixed per thread) ----
  const int sc2 = tid & 15;   // A column pair 0..15
  const int sr  = tid >> 4;   // A row base 0..7 (rows sr + 8*it)
  const int sn  = tid;        // B column 0..127

  // Token row pointers hoisted into registers (no LDS lookup in the k-loop).
  const float* aPtr[4];
#pragma unroll
  for (int it = 0; it < 4; ++it) {
    int t = token_list[e * TOKENS + rowTile * 32 + sr + 8 * it];
    aPtr[it] = x + (size_t)t * HIDDEN + 2 * sc2;
  }
  if (tid < 32) wts[tid] = weight_list[e * TOKENS + rowTile * 32 + tid];

  // Running B pointers: advance 2*INTER per packed pair -> walk all of K.
  const size_t wbase = (size_t)e * HIDDEN * INTER;
  const float* gCur = wg + wbase + nBase + sn;
  const float* uCur = wu + wbase + nBase + sn;

  unsigned* aStore  = &As32[sr * KP32 + sc2];  // + it*8*KP32 (imm offsets)
  unsigned* bgStore = &Bg32[sn * KP32];        // + k2       (imm offsets)
  unsigned* buStore = &Bu32[sn * KP32];

  // Fragment read pointers (constant per thread).
  const unsigned* aFrag  = &As32[(lane & 15) * KP32 + ((lane >> 4) << 3)];
  const unsigned* bgFrag = &Bg32[(wave * 32 + (lane & 15)) * KP32 + ((lane >> 4) << 3)];
  const unsigned* buFrag = &Bu32[(wave * 32 + (lane & 15)) * KP32 + ((lane >> 4) << 3)];

  v8f accg[2][2] = {};
  v8f accu[2][2] = {};

  for (int k0 = 0; k0 < HIDDEN; k0 += 32) {
    // ---- stage A: 4 independent float2 loads, imm-offset LDS stores ----
#pragma unroll
    for (int it = 0; it < 4; ++it) {
      float2 v = *reinterpret_cast<const float2*>(aPtr[it] + k0);
      aStore[it * 8 * KP32] = pack_bf16(v.x, v.y);
    }
    // ---- stage B: running pointers, 16 packed pairs per matrix ----
#pragma unroll
    for (int k2 = 0; k2 < 16; ++k2) {
      bgStore[k2] = pack_bf16(gCur[0], gCur[INTER]);
      buStore[k2] = pack_bf16(uCur[0], uCur[INTER]);
      gCur += 2 * INTER;
      uCur += 2 * INTER;
    }
    if (k0 + 32 < HIDDEN) {  // gCur/uCur now point at next k-step's row
      __builtin_prefetch(gCur, 0, 1);
      __builtin_prefetch(uCur, 0, 1);
    }
    __syncthreads();

    // ---- fragments: 2 x ds_load_b128 each ----
    v16bf a0  = FRAGP(aFrag, 0, 4);
    v16bf a1  = FRAGP(aFrag + 16 * KP32, 0, 4);
    v16bf bg0 = FRAGP(bgFrag, 0, 4);
    v16bf bg1 = FRAGP(bgFrag + 16 * KP32, 0, 4);
    v16bf bu0 = FRAGP(buFrag, 0, 4);
    v16bf bu1 = FRAGP(buFrag + 16 * KP32, 0, 4);

    accg[0][0] = __builtin_amdgcn_wmma_f32_16x16x32_bf16(false, a0, false, bg0, (short)0, accg[0][0], false, false);
    accg[0][1] = __builtin_amdgcn_wmma_f32_16x16x32_bf16(false, a0, false, bg1, (short)0, accg[0][1], false, false);
    accg[1][0] = __builtin_amdgcn_wmma_f32_16x16x32_bf16(false, a1, false, bg0, (short)0, accg[1][0], false, false);
    accg[1][1] = __builtin_amdgcn_wmma_f32_16x16x32_bf16(false, a1, false, bg1, (short)0, accg[1][1], false, false);
    accu[0][0] = __builtin_amdgcn_wmma_f32_16x16x32_bf16(false, a0, false, bu0, (short)0, accu[0][0], false, false);
    accu[0][1] = __builtin_amdgcn_wmma_f32_16x16x32_bf16(false, a0, false, bu1, (short)0, accu[0][1], false, false);
    accu[1][0] = __builtin_amdgcn_wmma_f32_16x16x32_bf16(false, a1, false, bu0, (short)0, accu[1][0], false, false);
    accu[1][1] = __builtin_amdgcn_wmma_f32_16x16x32_bf16(false, a1, false, bu1, (short)0, accu[1][1], false, false);
    __syncthreads();
  }

  // h = silu(g) * u * routing_weight -> bf16 hbuf (imm-offset stores)
  const int mrBase = (lane >> 4) << 3;
  bf16* hOut = hbuf + ((size_t)slotBase + mrBase) * INTER + nBase +
               wave * 32 + (lane & 15);
  const float* wRow = &wts[mrBase];
#pragma unroll
  for (int mt = 0; mt < 2; ++mt) {
#pragma unroll
    for (int j = 0; j < 8; ++j) {
      float wgt = wRow[mt * 16 + j];
#pragma unroll
      for (int nt = 0; nt < 2; ++nt) {
        float g = accg[mt][nt][j];
        float u = accu[mt][nt][j];
        float h = (g / (1.0f + __expf(-g))) * u * wgt;
        hOut[(size_t)(mt * 16 + j) * INTER + nt * 16] = (bf16)h;
      }
    }
  }
}

// ---------------------------------------------------------------------------
// Kernel 4: grouped down GEMM; scatter-accumulate into out with f32 atomics.
// Block = 128 threads (4 waves). Tile: 32 rows x 128 HIDDEN cols.
// ---------------------------------------------------------------------------
__global__ void __launch_bounds__(128) down_kernel(
    const float* __restrict__ wd, const bf16* __restrict__ hbuf,
    const int* __restrict__ token_list, const int* __restrict__ padded_counts,
    float* __restrict__ out) {
  const int e = blockIdx.z;
  const int rowTile = blockIdx.y;
  const int nBase = blockIdx.x * 128;
  const int pcnt = padded_counts[e];
  if (rowTile * 32 >= pcnt) return;
  int slotBase = 0;
  for (int i = 0; i < e; ++i) slotBase += padded_counts[i];
  slotBase += rowTile * 32;

  __shared__ unsigned As32[32 * KP32];   // hbuf rows, already bf16
  __shared__ unsigned Bs32[128 * KP32];  // w_down [n][k] bf16-packed
  __shared__ int tok[32];

  const int tid  = threadIdx.x;
  const int lane = tid & 31;
  const int wave = tid >> 5;

  const int sc2 = tid & 15;
  const int sr  = tid >> 4;
  const int sn  = tid;

  // A row pointers (u32 view of bf16 pairs), hoisted.
  const unsigned* aPtr[4];
#pragma unroll
  for (int it = 0; it < 4; ++it) {
    aPtr[it] = reinterpret_cast<const unsigned*>(
        hbuf + (size_t)(slotBase + sr + 8 * it) * INTER + 2 * sc2);
  }
  if (tid < 32) tok[tid] = token_list[e * TOKENS + rowTile * 32 + tid];

  const size_t wbase = (size_t)e * INTER * HIDDEN;
  const float* dCur = wd + wbase + nBase + sn;

  unsigned* aStore = &As32[sr * KP32 + sc2];
  unsigned* bStore = &Bs32[sn * KP32];

  const unsigned* aFrag = &As32[(lane & 15) * KP32 + ((lane >> 4) << 3)];
  const unsigned* bFrag = &Bs32[(wave * 32 + (lane & 15)) * KP32 + ((lane >> 4) << 3)];

  v8f acc[2][2] = {};

  for (int k0 = 0; k0 < INTER; k0 += 32) {
#pragma unroll
    for (int it = 0; it < 4; ++it) {
      aStore[it * 8 * KP32] = aPtr[it][k0 >> 1];
    }
#pragma unroll
    for (int k2 = 0; k2 < 16; ++k2) {
      bStore[k2] = pack_bf16(dCur[0], dCur[HIDDEN]);
      dCur += 2 * HIDDEN;
    }
    if (k0 + 32 < INTER) __builtin_prefetch(dCur, 0, 1);
    __syncthreads();

    v16bf a0 = FRAGP(aFrag, 0, 4);
    v16bf a1 = FRAGP(aFrag + 16 * KP32, 0, 4);
    v16bf b0 = FRAGP(bFrag, 0, 4);
    v16bf b1 = FRAGP(bFrag + 16 * KP32, 0, 4);

    acc[0][0] = __builtin_amdgcn_wmma_f32_16x16x32_bf16(false, a0, false, b0, (short)0, acc[0][0], false, false);
    acc[0][1] = __builtin_amdgcn_wmma_f32_16x16x32_bf16(false, a0, false, b1, (short)0, acc[0][1], false, false);
    acc[1][0] = __builtin_amdgcn_wmma_f32_16x16x32_bf16(false, a1, false, b0, (short)0, acc[1][0], false, false);
    acc[1][1] = __builtin_amdgcn_wmma_f32_16x16x32_bf16(false, a1, false, b1, (short)0, acc[1][1], false, false);
    __syncthreads();
  }

  const int mrBase = (lane >> 4) << 3;
  float* outCol = out + nBase + wave * 32 + (lane & 15);
  const int* tokRow = &tok[mrBase];
#pragma unroll
  for (int mt = 0; mt < 2; ++mt) {
#pragma unroll
    for (int j = 0; j < 8; ++j) {
      int t = tokRow[mt * 16 + j];
#pragma unroll
      for (int nt = 0; nt < 2; ++nt) {
        atomicAdd(outCol + (size_t)t * HIDDEN + nt * 16, acc[mt][nt][j]);
      }
    }
  }
}

// ---------------------------------------------------------------------------
extern "C" void kernel_launch(void* const* d_in, const int* in_sizes, int n_in,
                              void* d_out, int out_size, void* d_ws, size_t ws_size,
                              hipStream_t stream) {
  const float* x  = (const float*)d_in[0];  // [T, H]
  const float* wr = (const float*)d_in[1];  // [H, E]
  const float* wg = (const float*)d_in[2];  // [E, H, I]
  const float* wu = (const float*)d_in[3];  // [E, H, I]
  const float* wd = (const float*)d_in[4];  // [E, I, H]
  float* out = (float*)d_out;               // [T, H]

  char* p = (char*)d_ws;
  int*   topk_idx      = (int*)p;   p += (size_t)TOKENS * 2 * sizeof(int);
  float* topk_w        = (float*)p; p += (size_t)TOKENS * 2 * sizeof(float);
  int*   token_list    = (int*)p;   p += (size_t)NEXP * TOKENS * sizeof(int);
  float* weight_list   = (float*)p; p += (size_t)NEXP * TOKENS * sizeof(float);
  int*   padded_counts = (int*)p;   p += 64;
  bf16*  hbuf          = (bf16*)p;  // up to (2*TOKENS + 8*31) rows x INTER bf16

  zero_out_kernel<<<512, 256, 0, stream>>>(out, TOKENS * HIDDEN);
  router_kernel<<<TOKENS / 4, 128, 0, stream>>>(x, wr, topk_idx, topk_w);
  build_lists_kernel<<<NEXP, 32, 0, stream>>>(topk_idx, topk_w, token_list,
                                              weight_list, padded_counts);
  {
    dim3 grid(INTER / 128, TOKENS / 32, NEXP);
    gateup_kernel<<<grid, 128, 0, stream>>>(x, wg, wu, token_list, weight_list,
                                            padded_counts, hbuf);
  }
  {
    dim3 grid(HIDDEN / 128, TOKENS / 32, NEXP);
    down_kernel<<<grid, 128, 0, stream>>>(wd, hbuf, token_list, padded_counts, out);
  }
  (void)in_sizes; (void)n_in; (void)out_size; (void)ws_size;
}
